// StructureLoss_16527034155580
// MI455X (gfx1250) — matrix operand order repaired and checked by reading, not compiled, not run on hardware
//
#include <hip/hip_runtime.h>

typedef __attribute__((ext_vector_type(16))) _Float16 v16h;
typedef __attribute__((ext_vector_type(8)))  float    v8f;

#define NB   16
#define CCH  2
#define HH   512
#define WW   512
#define KW   31
#define PADW 15
#define TILE 32            // output tile per block
#define HALO 62            // TILE + 2*PADW
#define NTY  (HH / TILE)   // 16
#define NTX  (WW / TILE)   // 16
#define NBLK (NB * NTY * NTX) // 4096 blocks, 5 floats of partials each

// A-fragment (16-bit, 16xK=32) element -> K index, per ISA 7.12.2:
// lanes 0-15: VGPR0..3 hold K=0..7 (pairs), VGPR4..7 hold K=16..23
// lanes 16-31: same +8.  e = element index 0..15 within v16h.
__device__ __forceinline__ int a_k_of(int e, int hi) {
  int v = e >> 1, p = e & 1;
  int base = (v < 4) ? (2 * v) : (2 * (v - 4) + 16);
  return base + p + 8 * hi; // K in [0,31]
}
// B-fragment (Kx16): lane holds column n = lane&15; element e -> K = a_k_of(e, hi), same order as A.
// C/D fragment (16x16 f32): lane holds column n = lane&15; VGPR v -> row m = v + 8*hi.

__device__ __forceinline__ int clamp63(int i) { return i > 63 ? 63 : i; }

__global__ void __launch_bounds__(128)
structure_loss_main(const float* __restrict__ pred,
                    const int*   __restrict__ target,
                    float*       __restrict__ parts)
{
  // Both LDS arrays are written over their FULL extent (zeros outside the valid
  // 62x62 / 62x32 region), so fragment fills never need divergent guards: any
  // index that can exceed 63 is clamped to 63, which always holds 0.0, and the
  // banded partner matrix is 0 at every K where the clamp can trigger.
  __shared__ _Float16 smask[64][64]; // mask0 halo
  __shared__ _Float16 shs[64][34];   // horizontal 31-wide sums (<=31, exact in f16)
  __shared__ float    swred[4][5];

  const int tid  = threadIdx.x;
  const int lane = tid & 31;
  const int wv   = tid >> 5;     // wave 0..3
  const int hi   = lane >> 4;    // lane half
  const int lm   = lane & 15;

  const int txb = blockIdx.x * TILE;
  const int tyb = blockIdx.y * TILE;
  const int img = blockIdx.z;

  // ---- stage mask0 halo into LDS (zero outside image / outside 62x62) ----
  const long ibase = (long)img * (HH * WW);
  for (int idx = tid; idx < 64 * 64; idx += 128) {
    int r = idx >> 6, c = idx & 63;
    float m0 = 0.0f;
    int y = tyb - PADW + r;
    int x = txb - PADW + c;
    if (r < HALO && c < HALO && (unsigned)y < (unsigned)HH && (unsigned)x < (unsigned)WW)
      m0 = (target[ibase + (long)y * WW + x] == 0) ? 1.0f : 0.0f;
    smask[r][c] = (_Float16)m0;
  }
  __syncthreads();

  // ---- horizontal pass: wave wv -> Hs rows [wv*16, wv*16+16), col subtiles cs=0,1
  //      Hs[r][j] = sum_{k=j..j+30} smask[r][cs*16+k]  (banded-ones on B side)
  for (int cs = 0; cs < 2; ++cs) {
    v16h a1, a2, b1, b2;
    const int r = wv * 16 + lm;        // A row for this lane (<=63, in-bounds)
#pragma unroll
    for (int e = 0; e < 16; ++e) {
      int ka = a_k_of(e, hi);
      int c1 = cs * 16 + ka;           // <= 47: always in-bounds
      int c2 = clamp63(c1 + 32);       // clamp hits only where B2 == 0
      a1[e] = smask[r][c1];
      a2[e] = smask[r][c2];
      int kb = a_k_of(e, hi);          // K of B fragment
      int j  = lm;                     // column
      b1[e] = (kb >= j && kb <= j + 30) ? (_Float16)1.0f : (_Float16)0.0f;
      b2[e] = (kb + 2 <= j)             ? (_Float16)1.0f : (_Float16)0.0f; // K' = kb+32
    }
    v8f acc = {};
    acc = __builtin_amdgcn_wmma_f32_16x16x32_f16(false, a1, false, b1, (short)0, acc, false, false);
    acc = __builtin_amdgcn_wmma_f32_16x16x32_f16(false, a2, false, b2, (short)0, acc, false, false);
    // Rows 62/63 (wave 3) are computed from zeroed smask rows -> exactly 0.0,
    // so the store is unconditional (no exec-mask juggling).
#pragma unroll
    for (int v = 0; v < 8; ++v)
      shs[wv * 16 + v + 8 * hi][cs * 16 + lm] = (_Float16)acc[v];
  }
  __syncthreads();

  // ---- vertical pass: wave quadrant (qy,qx); banded-ones on A side
  const int qy = wv >> 1, qx = wv & 1;
  v16h va1, va2, vb1, vb2;
#pragma unroll
  for (int e = 0; e < 16; ++e) {
    int ka = a_k_of(e, hi);
    int i  = lm; // output row within quadrant
    va1[e] = (ka >= i && ka <= i + 30) ? (_Float16)1.0f : (_Float16)0.0f;
    va2[e] = (ka + 2 <= i)             ? (_Float16)1.0f : (_Float16)0.0f; // K' = ka+32
    int kb = a_k_of(e, hi);
    int r1 = qy * 16 + kb;             // <= 47: always in-bounds
    int r2 = clamp63(r1 + 32);         // clamp hits only where A2 == 0
    int c  = qx * 16 + lm;
    vb1[e] = shs[r1][c];
    vb2[e] = shs[r2][c];
  }
  v8f cnt = {};
  cnt = __builtin_amdgcn_wmma_f32_16x16x32_f16(false, va1, false, vb1, (short)0, cnt, false, false);
  cnt = __builtin_amdgcn_wmma_f32_16x16x32_f16(false, va2, false, vb2, (short)0, cnt, false, false);
  // cnt[v] = exact 31x31 zero-padded box count of mask0 at this pixel.

  // ---- elementwise phase: BCE + weighted inter/union for both channels ----
  float accB = 0.f, accI0 = 0.f, accI1 = 0.f, accU0 = 0.f, accU1 = 0.f;
  const long pbase0 = ((long)img * CCH + 0) * (long)(HH * WW);
  const long pbase1 = ((long)img * CCH + 1) * (long)(HH * WW);
  const float inv = 1.0f / (float)(KW * KW);
#pragma unroll
  for (int v = 0; v < 8; ++v) {
    int m = v + 8 * hi;
    int y = tyb + qy * 16 + m;
    int x = txb + qx * 16 + lm;
    float mask0 = (float)smask[qy * 16 + m + PADW][qx * 16 + lm + PADW];
    float mask1 = 1.0f - mask0;
    float c0 = cnt[v];
    int yl = y - PADW; if (yl < 0) yl = 0;
    int yh = y + PADW; if (yh > HH - 1) yh = HH - 1;
    int xl = x - PADW; if (xl < 0) xl = 0;
    int xh = x + PADW; if (xh > WW - 1) xh = WW - 1;
    float cOnes = (float)((yh - yl + 1) * (xh - xl + 1)); // box count of all-ones
    float bm0 = c0 * inv;
    float bm1 = (cOnes - c0) * inv;                       // box_mean(mask1)
    float w0 = 1.0f + 5.0f * fabsf(bm0 - mask0);
    float w1 = 1.0f + 5.0f * fabsf(bm1 - mask1);
    long off = (long)y * WW + x;
    float p0 = pred[pbase0 + off];
    float p1 = pred[pbase1 + off];
    float e0 = __expf(-fabsf(p0));
    float e1 = __expf(-fabsf(p1));
    accB += fmaxf(p0, 0.f) - p0 * mask0 + __logf(1.0f + e0)
          + fmaxf(p1, 0.f) - p1 * mask1 + __logf(1.0f + e1);
    float s0 = 1.0f / (1.0f + __expf(-p0));
    float s1 = 1.0f / (1.0f + __expf(-p1));
    accI0 += s0 * mask0 * w0;  accU0 += (s0 + mask0) * w0;
    accI1 += s1 * mask1 * w1;  accU1 += (s1 + mask1) * w1;
  }

  // ---- deterministic block reduction: wave shuffle tree, then fixed 4-way sum ----
  float vals[5] = {accI0, accI1, accU0, accU1, accB};
#pragma unroll
  for (int q = 0; q < 5; ++q) {
    float s = vals[q];
#pragma unroll
    for (int o = 16; o > 0; o >>= 1) s += __shfl_xor(s, o, 32);
    if (lane == 0) swred[wv][q] = s;
  }
  __syncthreads();
  if (tid == 0) {
    long flat = ((long)img * NTY + blockIdx.y) * NTX + blockIdx.x;
#pragma unroll
    for (int q = 0; q < 5; ++q)
      parts[flat * 5 + q] = swred[0][q] + swred[1][q] + swred[2][q] + swred[3][q];
  }
}

__global__ void __launch_bounds__(256)
structure_loss_finalize(const float* __restrict__ parts, float* __restrict__ out)
{
  __shared__ float red[256];
  __shared__ float acc[16][4];
  __shared__ float bceTot;
  const int tid = threadIdx.x;
  const int per = NTY * NTX; // 256 blocks per image == blockDim

  for (int nq = 0; nq < 16 * 4; ++nq) {
    int n = nq >> 2, q = nq & 3;
    red[tid] = parts[(long)(n * per + tid) * 5 + q];
    __syncthreads();
    for (int st = 128; st > 0; st >>= 1) {
      if (tid < st) red[tid] += red[tid + st];
      __syncthreads();
    }
    if (tid == 0) acc[n][q] = red[0];
    __syncthreads();
  }
  {
    float s = 0.f;
    for (int i = tid; i < NBLK; i += 256) s += parts[(long)i * 5 + 4];
    red[tid] = s;
    __syncthreads();
    for (int st = 128; st > 0; st >>= 1) {
      if (tid < st) red[tid] += red[tid + st];
      __syncthreads();
    }
    if (tid == 0) bceTot = red[0];
    __syncthreads();
  }
  if (tid == 0) {
    float wbce = bceTot / (float)((long)NB * CCH * HH * WW);
    float loss = 0.f;
    for (int n = 0; n < 16; ++n)
      for (int c = 0; c < 2; ++c) {
        float inter = acc[n][c];
        float uni   = acc[n][2 + c];
        float wiou  = 1.0f - (inter + 1.0f) / (uni - inter + 1.0f);
        loss += wbce + wiou;
      }
    out[0] = loss / 32.0f;
  }
}

extern "C" void kernel_launch(void* const* d_in, const int* in_sizes, int n_in,
                              void* d_out, int out_size, void* d_ws, size_t ws_size,
                              hipStream_t stream) {
  const float* pred   = (const float*)d_in[0];
  const int*   target = (const int*)d_in[1];
  float* parts = (float*)d_ws; // NBLK * 5 floats = 80 KB, fully rewritten every launch
  dim3 grid(NTX, NTY, NB);
  structure_loss_main<<<grid, 128, 0, stream>>>(pred, target, parts);
  structure_loss_finalize<<<1, 256, 0, stream>>>(parts, (float*)d_out);
}
